// A_SPN_89756226552165
// MI455X (gfx1250) — compile-verified, matrix operand order checked by reading, not run on hardware
//
#include <hip/hip_runtime.h>

// ---------------------------------------------------------------------------
// Dims
// ---------------------------------------------------------------------------
#define BS   64
#define C    270
#define HW   729
#define NCLS 16
#define HWP  768      // hw padded (mult of 128)
#define CKP  288      // C padded for K dim (mult of 32)
#define CMP  320      // C padded for M dim (mult of 64)
#define CENTER 364    // (HW-1)/2

typedef _Float16 v16h __attribute__((ext_vector_type(16)));
typedef _Float16 v8h  __attribute__((ext_vector_type(8)));
typedef float    v8f  __attribute__((ext_vector_type(8)));

// ---------------------------------------------------------------------------
// WMMA helpers (CDNA5 wave32, 16x16x32 f16 -> f32)
// ---------------------------------------------------------------------------
__device__ __forceinline__ v8f wmma16(v16h a, v16h b, v8f c) {
  return __builtin_amdgcn_wmma_f32_16x16x32_f16(false, a, false, b, (short)0, c,
                                                false, false);
}

// Per-lane fragment load for a row-major [rows x K] f16 matrix (A and B^T both
// use the identical "row" pattern on CDNA5):
//   lane L: row = L&15, K-chunks {kb..kb+7, kb+16..kb+23}, kb = (L>>4)*8.
__device__ __forceinline__ v16h ldfrag(const _Float16* p) {
  v8h lo = *(const v8h*)(p);
  v8h hi = *(const v8h*)(p + 16);
  return __builtin_shufflevector(lo, hi, 0, 1, 2, 3, 4, 5, 6, 7,
                                 8, 9, 10, 11, 12, 13, 14, 15);
}

// ---------------------------------------------------------------------------
// Wave-level matrix tiles: A(MxK) * B^T, both row-major along K.
// ---------------------------------------------------------------------------
struct Acc4 { v8f a00, a01, a10, a11; };            // 32x32 tile (2x2)

__device__ __forceinline__ Acc4 wave_mm(const _Float16* A, const _Float16* Bt,
                                        int lda, int ldb, int ksteps) {
  const int lane = threadIdx.x & 31;
  const int r    = lane & 15;
  const int hi8  = (lane >> 4) * 8;
  const _Float16* Ap = A + (size_t)r * lda + hi8;
  const _Float16* Bp = Bt + (size_t)r * ldb + hi8;
  v8f z = {0.f, 0.f, 0.f, 0.f, 0.f, 0.f, 0.f, 0.f};
  Acc4 s; s.a00 = z; s.a01 = z; s.a10 = z; s.a11 = z;
  for (int k = 0; k < ksteps; ++k) {
    v16h a0 = ldfrag(Ap);
    v16h a1 = ldfrag(Ap + 16 * (size_t)lda);
    v16h b0 = ldfrag(Bp);
    v16h b1 = ldfrag(Bp + 16 * (size_t)ldb);
    s.a00 = wmma16(a0, b0, s.a00);
    s.a01 = wmma16(a0, b1, s.a01);
    s.a10 = wmma16(a1, b0, s.a10);
    s.a11 = wmma16(a1, b1, s.a11);
    Ap += 32; Bp += 32;
  }
  return s;
}

struct Acc8 { v8f a[2][4]; };                       // 32x64 tile (2x4)

__device__ __forceinline__ Acc8 wave_mm_2x4(const _Float16* A, const _Float16* Bt,
                                            int lda, int ldb, int ksteps) {
  const int lane = threadIdx.x & 31;
  const int r    = lane & 15;
  const int hi8  = (lane >> 4) * 8;
  const _Float16* Ap = A + (size_t)r * lda + hi8;
  const _Float16* Bp = Bt + (size_t)r * ldb + hi8;
  v8f z = {0.f, 0.f, 0.f, 0.f, 0.f, 0.f, 0.f, 0.f};
  Acc8 s;
#pragma unroll
  for (int i = 0; i < 2; ++i)
#pragma unroll
    for (int j = 0; j < 4; ++j) s.a[i][j] = z;
  for (int k = 0; k < ksteps; ++k) {
    v16h a0 = ldfrag(Ap);
    v16h a1 = ldfrag(Ap + 16 * (size_t)lda);
    v16h b0 = ldfrag(Bp);
    v16h b1 = ldfrag(Bp + 16 * (size_t)ldb);
    v16h b2 = ldfrag(Bp + 32 * (size_t)ldb);
    v16h b3 = ldfrag(Bp + 48 * (size_t)ldb);
    s.a[0][0] = wmma16(a0, b0, s.a[0][0]);
    s.a[0][1] = wmma16(a0, b1, s.a[0][1]);
    s.a[0][2] = wmma16(a0, b2, s.a[0][2]);
    s.a[0][3] = wmma16(a0, b3, s.a[0][3]);
    s.a[1][0] = wmma16(a1, b0, s.a[1][0]);
    s.a[1][1] = wmma16(a1, b1, s.a[1][1]);
    s.a[1][2] = wmma16(a1, b2, s.a[1][2]);
    s.a[1][3] = wmma16(a1, b3, s.a[1][3]);
    Ap += 32; Bp += 32;
  }
  return s;
}

// ---------------------------------------------------------------------------
// K1: BatchNorm statistics (sum / sumsq per channel), coalesced over C
// ---------------------------------------------------------------------------
__global__ void k_bnstats(const float* __restrict__ x, float* macc, float* sacc) {
  const int c = threadIdx.x;
  if (c >= C) return;
  const int r0 = blockIdx.x * 183;
  const int r1 = min(r0 + 183, BS * HW);
  float s = 0.f, s2 = 0.f;
  for (int row = r0; row < r1; ++row) {
    float v = x[(size_t)row * C + c];
    s += v; s2 += v * v;
  }
  atomicAdd(&macc[c], s);
  atomicAdd(&sacc[c], s2);
}

__global__ void k_bnfinal(const float* macc, const float* sacc,
                          const float* gamma, const float* beta,
                          float* scale, float* shift) {
  const int c = threadIdx.x;
  if (c >= C) return;
  const float inv_n = 1.f / (float)(BS * HW);
  float mean = macc[c] * inv_n;
  float var  = sacc[c] * inv_n - mean * mean;
  float sc   = gamma[c] * rsqrtf(var + 1e-5f);
  scale[c] = sc;
  shift[c] = beta[c] - mean * sc;
}

// ---------------------------------------------------------------------------
// K2: per-(b,c) row sum-of-squares of BN'ed data, then write xn (f16, 2 layouts)
// ---------------------------------------------------------------------------
__global__ void k_rowsq(const float* __restrict__ x, const float* scale,
                        const float* shift, float* rsum2) {
  const int c = threadIdx.x;
  if (c >= C) return;
  const int b = blockIdx.x;
  const int i0 = blockIdx.y * 183, i1 = min(i0 + 183, HW);
  const float sc = scale[c], sh = shift[c];
  float acc = 0.f;
  for (int i = i0; i < i1; ++i) {
    float v = x[((size_t)b * HW + i) * C + c] * sc + sh;
    acc += v * v;
  }
  atomicAdd(&rsum2[b * C + c], acc);
}

__global__ void k_rowsq_inv(const float* rsum2, float* rinv) {
  int idx = blockIdx.x * blockDim.x + threadIdx.x;
  if (idx < BS * C) rinv[idx] = rsqrtf(fmaxf(rsum2[idx], 1e-20f));
}

__global__ void k_writexn(const float* __restrict__ x, const float* scale,
                          const float* shift, const float* rinv,
                          _Float16* __restrict__ xnT, _Float16* __restrict__ xn) {
  const int c = threadIdx.x;
  if (c >= C) return;
  const int b = blockIdx.x;
  const int i0 = blockIdx.y * 122, i1 = min(i0 + 122, HW);
  const float sc = scale[c], sh = shift[c], rv = rinv[b * C + c];
  for (int i = i0; i < i1; ++i) {
    float v = (x[((size_t)b * HW + i) * C + c] * sc + sh) * rv;
    xnT[((size_t)b * HWP + i) * CKP + c] = (_Float16)v;   // [b, hw, c] for GEMM1
    xn [((size_t)b * CMP + c) * HWP + i] = (_Float16)v;   // [b, c, hw] for GEMM2
  }
}

// ---------------------------------------------------------------------------
// K3: GEMM1 fused  (Xmm = xnT * xnT^T, never stored)
//   per-row: r2 = sum_j Xmm^2, tt = sum_j Xmm*kernel, central row written out
//   block tile 64x128 (4 waves, each 32x64 = 2x4 WMMA)
// ---------------------------------------------------------------------------
__global__ __launch_bounds__(128)
void k_gemm1(const _Float16* __restrict__ xnT, const float* __restrict__ kern,
             float* __restrict__ r2, float* __restrict__ tt,
             float* __restrict__ central) {
  const int b    = blockIdx.z;
  const int wave = threadIdx.x >> 5;
  const int M0   = blockIdx.x * 64 + (wave >> 1) * 32;
  const int N0   = blockIdx.y * 128 + (wave & 1) * 64;
  const _Float16* base = xnT + (size_t)b * HWP * CKP;
  Acc8 s = wave_mm_2x4(base + (size_t)M0 * CKP, base + (size_t)N0 * CKP,
                       CKP, CKP, CKP / 32);
  const int lane = threadIdx.x & 31, r = lane & 15, hi = lane >> 4;
  float kw[4];
#pragma unroll
  for (int nt = 0; nt < 4; ++nt) {
    const int j = N0 + nt * 16 + r;
    kw[nt] = (j < HW) ? kern[b * HW + j] : 0.f;
  }
#pragma unroll
  for (int mt = 0; mt < 2; ++mt) {
#pragma unroll
    for (int v = 0; v < 8; ++v) {
      const int gm = M0 + mt * 16 + hi * 8 + v;
      float sq = 0.f, tv = 0.f;
#pragma unroll
      for (int nt = 0; nt < 4; ++nt) {
        float xv = s.a[mt][nt][v];
        if (gm == CENTER) central[b * HWP + N0 + nt * 16 + r] = xv;
        sq += xv * xv;
        tv += xv * kw[nt];
      }
#pragma unroll
      for (int off = 1; off < 16; off <<= 1) {   // reduce over the 16-lane half
        sq += __shfl_xor(sq, off, 32);
        tv += __shfl_xor(tv, off, 32);
      }
      if (r == 0) {
        atomicAdd(&r2[b * HWP + gm], sq);
        atomicAdd(&tt[b * HWP + gm], tv);
      }
    }
  }
}

// ---------------------------------------------------------------------------
// K4a: g = central / ||row_center||, plus per-batch gmax/gmin
// ---------------------------------------------------------------------------
__global__ void k_central(const float* central, const float* r2, float* g,
                          float* gmaxA, float* gminA) {
  const int b = blockIdx.x;
  __shared__ float smax[256], smin[256];
  const float rinv = rsqrtf(fmaxf(r2[b * HWP + CENTER], 1e-30f));
  float mx = -1e30f, mn = 1e30f;
  for (int k = threadIdx.x; k < HW; k += 256) {
    float gg = central[b * HWP + k] * rinv;
    g[b * HWP + k] = gg;
    mx = fmaxf(mx, gg); mn = fminf(mn, gg);
  }
  smax[threadIdx.x] = mx; smin[threadIdx.x] = mn;
  __syncthreads();
  for (int s = 128; s > 0; s >>= 1) {
    if (threadIdx.x < s) {
      smax[threadIdx.x] = fmaxf(smax[threadIdx.x], smax[threadIdx.x + s]);
      smin[threadIdx.x] = fminf(smin[threadIdx.x], smin[threadIdx.x + s]);
    }
    __syncthreads();
  }
  if (threadIdx.x == 0) { gmaxA[b] = smax[0]; gminA[b] = smin[0]; }
}

// ---------------------------------------------------------------------------
// K4b: per (b,i): s_i = tt/||row||, rowmax m_i = s_i*gmax|gmin, Z_i (softmax)
//   (rank-1 structure: logits row i = s_i * g + const  ->  bias cancels)
// ---------------------------------------------------------------------------
__global__ void k_zrow(const float* r2, const float* tt, const float* g,
                       const float* gmaxA, const float* gminA,
                       float* sarr, float* mrow, float* zinv) {
  const int i = blockIdx.x, b = blockIdx.y;
  __shared__ float red[128];
  const float si = tt[b * HWP + i] * rsqrtf(fmaxf(r2[b * HWP + i], 1e-30f));
  const float m  = (si >= 0.f) ? si * gmaxA[b] : si * gminA[b];
  float p = 0.f;
  for (int k = threadIdx.x; k < HW; k += 128)
    p += __expf(si * g[b * HWP + k] - m);
  red[threadIdx.x] = p;
  __syncthreads();
  for (int s = 64; s > 0; s >>= 1) {
    if (threadIdx.x < s) red[threadIdx.x] += red[threadIdx.x + s];
    __syncthreads();
  }
  if (threadIdx.x == 0) {
    sarr[b * HWP + i] = si;
    mrow[b * HWP + i] = m;
    zinv[b * HWP + i] = 1.f / red[0];
  }
}

// ---------------------------------------------------------------------------
// K5: materialize att^T in f16:  attT[b,k,i] = exp(s_i*g_k - m_i)*zinv_i
// ---------------------------------------------------------------------------
__global__ void k_att(const float* sarr, const float* mrow, const float* zinv,
                      const float* g, _Float16* __restrict__ attT) {
  const int k = blockIdx.x, b = blockIdx.y;
  const float gk = g[b * HWP + k];
  _Float16* row = attT + ((size_t)b * HWP + k) * HWP;
  for (int i = threadIdx.x; i < HWP; i += 256) {
    float v = 0.f;
    if (i < HW)
      v = __expf(sarr[b * HWP + i] * gk - mrow[b * HWP + i]) * zinv[b * HWP + i];
    row[i] = (_Float16)v;
  }
}

// ---------------------------------------------------------------------------
// K6: GEMM2  aout[c,k] = sum_h xn[c,h] * attT[k,h]   (A*B^T), f16 out
//   block tile 64x128 (4 waves, each 32x64)
// ---------------------------------------------------------------------------
__global__ __launch_bounds__(128)
void k_gemm2(const _Float16* __restrict__ xn, const _Float16* __restrict__ attT,
             _Float16* __restrict__ aout) {
  const int b    = blockIdx.z;
  const int wave = threadIdx.x >> 5;
  const int M0   = blockIdx.x * 64 + (wave >> 1) * 32;
  const int N0   = blockIdx.y * 128 + (wave & 1) * 64;
  Acc8 s = wave_mm_2x4(xn + (size_t)b * CMP * HWP + (size_t)M0 * HWP,
                       attT + (size_t)b * HWP * HWP + (size_t)N0 * HWP,
                       HWP, HWP, HWP / 32);
  const int lane = threadIdx.x & 31, r = lane & 15, hi = lane >> 4;
  _Float16* out = aout + (size_t)b * CMP * HWP;
#pragma unroll
  for (int mt = 0; mt < 2; ++mt)
#pragma unroll
    for (int nt = 0; nt < 4; ++nt)
#pragma unroll
      for (int v = 0; v < 8; ++v) {
        const int gm = M0 + mt * 16 + hi * 8 + v;
        const int gn = N0 + nt * 16 + r;
        out[(size_t)gm * HWP + gn] = (_Float16)s.a[mt][nt][v];
      }
}

// ---------------------------------------------------------------------------
// K7: GEMM3  sop = aout * aout^T, f32 out (small; 64x64 block tile, 2x2 waves)
// ---------------------------------------------------------------------------
__global__ __launch_bounds__(128)
void k_gemm3(const _Float16* __restrict__ aout, float* __restrict__ sop) {
  const int b    = blockIdx.z;
  const int wave = threadIdx.x >> 5;
  const int M0   = blockIdx.x * 64 + (wave >> 1) * 32;
  const int N0   = blockIdx.y * 64 + (wave & 1) * 32;
  const _Float16* base = aout + (size_t)b * CMP * HWP;
  Acc4 s = wave_mm(base + (size_t)M0 * HWP, base + (size_t)N0 * HWP, HWP, HWP,
                   HWP / 32);
  const int lane = threadIdx.x & 31, r = lane & 15, hi = lane >> 4;
  float* out = sop + (size_t)b * CMP * CMP;
  v8f accs[4] = {s.a00, s.a01, s.a10, s.a11};
#pragma unroll
  for (int mt = 0; mt < 2; ++mt)
#pragma unroll
    for (int nt = 0; nt < 2; ++nt)
#pragma unroll
      for (int v = 0; v < 8; ++v) {
        const int gm = M0 + mt * 16 + hi * 8 + v;
        const int gn = N0 + nt * 16 + r;
        out[(size_t)gm * CMP + gn] = accs[mt * 2 + nt][v];
      }
}

// ---------------------------------------------------------------------------
// K8: row L2 norms of sop (double-l2norm == single norm)
// ---------------------------------------------------------------------------
__global__ void k_soprn(const float* __restrict__ sop, float* rninv) {
  const int idx = blockIdx.x * blockDim.x + threadIdx.x;
  if (idx >= BS * C) return;
  const int b = idx / C, c1 = idx % C;
  const float* row = sop + ((size_t)b * CMP + c1) * CMP;
  float sum = 0.f;
  for (int c2 = 0; c2 < C; ++c2) { float v = row[c2]; sum += v * v; }
  rninv[idx] = rsqrtf(fmaxf(sum, 1e-30f));
}

// ---------------------------------------------------------------------------
// K9: final FC  out[b,n] = sum sop_norm * W + bias
// ---------------------------------------------------------------------------
__global__ void k_fc(const float* __restrict__ sop, const float* __restrict__ rninv,
                     const float* __restrict__ fcw, const float* __restrict__ fcb,
                     float* __restrict__ out) {
  const int n = blockIdx.x, b = blockIdx.y;
  __shared__ float red[256];
  const float* W = fcw + (size_t)n * (C * C);
  float p = 0.f;
  for (int idx = threadIdx.x; idx < C * C; idx += 256) {
    const int c1 = idx / C, c2 = idx - c1 * C;
    p += sop[((size_t)b * CMP + c1) * CMP + c2] * rninv[b * C + c1] * W[idx];
  }
  red[threadIdx.x] = p;
  __syncthreads();
  for (int s = 128; s > 0; s >>= 1) {
    if (threadIdx.x < s) red[threadIdx.x] += red[threadIdx.x + s];
    __syncthreads();
  }
  if (threadIdx.x == 0) out[b * NCLS + n] = red[0] + fcb[n];
}

// ---------------------------------------------------------------------------
// Launch
// ---------------------------------------------------------------------------
extern "C" void kernel_launch(void* const* d_in, const int* in_sizes, int n_in,
                              void* d_out, int out_size, void* d_ws, size_t ws_size,
                              hipStream_t stream) {
  const float* x     = (const float*)d_in[0];
  const float* gamma = (const float*)d_in[1];
  const float* beta  = (const float*)d_in[2];
  const float* kern  = (const float*)d_in[3];
  // d_in[4] = bias: constant over softmax axis -> cancels; unused.
  const float* fcw   = (const float*)d_in[5];
  const float* fcb   = (const float*)d_in[6];
  float* out = (float*)d_out;

  // ---- workspace layout (f32 region then f16 region) ----
  float* f = (float*)d_ws;
  float* scale  = f;            // 288
  float* shift  = f + 288;      // 288
  float* macc   = f + 576;      // 288  (zeroed)
  float* sacc   = f + 864;      // 288  (zeroed)
  float* rsum2  = f + 1152;     // 17280 (zeroed)
  float* rinv   = f + 18432;    // 17280
  float* r2     = f + 35712;    // 64*768 (zeroed)
  float* tt     = f + 84864;    // 64*768 (partly zeroed)
  float* central= f + 134016;   // 64*768
  float* g      = f + 183168;   // 64*768
  float* sarr   = f + 232320;   // 64*768
  float* mrow   = f + 281472;   // 64*768
  float* zinv   = f + 330624;   // 64*768
  float* gmaxA  = f + 379776;   // 64
  float* gminA  = f + 379840;   // 64
  float* rninv  = f + 379904;   // 17280
  float* sop    = f + 397184;   // 64*320*320
  _Float16* h   = (_Float16*)(f + 397184 + (size_t)BS * CMP * CMP);
  _Float16* xnT = h;                                      // 64*768*288 (zeroed)
  _Float16* xn  = xnT + (size_t)BS * HWP * CKP;           // 64*320*768 (zeroed)
  _Float16* aout= xn  + (size_t)BS * CMP * HWP;           // 64*320*768
  _Float16* attT= aout + (size_t)BS * CMP * HWP;          // 64*768*768 (zeroed)

  // ---- zero the accumulator / padded buffers (required every call) ----
  hipMemsetAsync(macc, 0, (size_t)(288 + 288 + 17280 + 2 * 64 * HWP) * 4, stream);
  hipMemsetAsync(xnT, 0, ((size_t)BS * HWP * CKP + (size_t)BS * CMP * HWP) * 2, stream);
  hipMemsetAsync(attT, 0, (size_t)BS * HWP * HWP * 2, stream);

  // ---- BN stats + normalize ----
  k_bnstats<<<255, 288, 0, stream>>>(x, macc, sacc);
  k_bnfinal<<<1, 288, 0, stream>>>(macc, sacc, gamma, beta, scale, shift);
  k_rowsq<<<dim3(BS, 4), 288, 0, stream>>>(x, scale, shift, rsum2);
  k_rowsq_inv<<<(BS * C + 255) / 256, 256, 0, stream>>>(rsum2, rinv);
  k_writexn<<<dim3(BS, 6), 288, 0, stream>>>(x, scale, shift, rinv, xnT, xn);

  // ---- GEMM1 fused (Xmm never materialized) ----
  k_gemm1<<<dim3(HWP / 64, HWP / 128, BS), 128, 0, stream>>>(xnT, kern, r2, tt,
                                                             central);

  // ---- rank-1 softmax pieces ----
  k_central<<<BS, 256, 0, stream>>>(central, r2, g, gmaxA, gminA);
  k_zrow<<<dim3(HW, BS), 128, 0, stream>>>(r2, tt, g, gmaxA, gminA, sarr, mrow, zinv);
  k_att<<<dim3(HW, BS), 256, 0, stream>>>(sarr, mrow, zinv, g, attT);

  // ---- GEMM2: aout = xn * att ----
  k_gemm2<<<dim3(CMP / 64, HWP / 128, BS), 128, 0, stream>>>(xn, attT, aout);

  // ---- GEMM3: sop = aout * aout^T ----
  k_gemm3<<<dim3(CMP / 64, CMP / 64, BS), 128, 0, stream>>>(aout, sop);

  // ---- normalize + FC ----
  k_soprn<<<(BS * C + 255) / 256, 256, 0, stream>>>(sop, rninv);
  k_fc<<<dim3(NCLS, BS), 256, 0, stream>>>(sop, rninv, fcw, fcb, out);
}
